// TrafficSeq2Seq_44839458570911
// MI455X (gfx1250) — matrix operand order repaired and checked by reading, not compiled, not run on hardware
//
#include <hip/hip_runtime.h>
#include <hip/hip_bf16.h>

// ---------------------------------------------------------------------------
// TrafficSeq2Seq (GRU encoder/decoder + attention) for MI455X / gfx1250.
// All GEMMs on v_wmma_f32_16x16x32_bf16 (bf16 operands, f32 accumulate).
// Activations kept in bf16 alongside the f32 hidden chain: WMMA A-fragments
// are raw b128 loads. All GEMM N dims padded to an EVEN tile count so the
// inner loop is branch-free (no EXEC manipulation around WMMA, per ISA 7.12).
// ---------------------------------------------------------------------------

typedef __attribute__((ext_vector_type(16))) __bf16 v16bf;
typedef __attribute__((ext_vector_type(8)))  float  v8f;

#define B_      128
#define TIN_    256
#define SOUT_   32
#define H_      512
#define THREEH_ 1536
#define IND_    512
#define D3_     300       // decoder out dim (3*OUT_DIM)
#define D3S_    304       // store limit padded to 16
#define D3N_    320       // N padded to EVEN tile count (20 tiles)
#define D3K_    320       // K depth padded to 32
#define CAT_    812
#define CATP_   832       // K padded to 32

__device__ __forceinline__ float sigmoidf_(float x) {
    return 1.f / (1.f + __expf(-x));
}
__device__ __forceinline__ float tanh_fast_(float x) {
    float e = __expf(2.f * x);
    return (e - 1.f) / (e + 1.f);
}

// A fragment 16x32 bf16: lane<16 holds K 0-15 of row (lane&15); lane>=16
// holds K 16-31 (matches load_b_w). One contiguous 32B chunk.
__device__ __forceinline__ v16bf load_a_bf(const __bf16* __restrict__ A, int ldA,
                                           int mbase, int k, int lane) {
    int row = mbase + (lane & 15);
    int hi  = lane >> 4;
    const __bf16* p = A + (size_t)row * ldA + k + 16 * hi;
    v16bf a;
#pragma unroll
    for (int i = 0; i < 16; ++i) a[i] = p[i];
    return a;
}

// B fragment 32x16 bf16 (K x N): lane owns one N-column = one row-major row of
// W (N x Kpad), 16 contiguous K values -> one 32B contiguous read.
__device__ __forceinline__ v16bf load_b_w(const __bf16* __restrict__ W, int ldW,
                                          int nbase, int k, int lane) {
    const __bf16* p = W + (size_t)(nbase + (lane & 15)) * ldW + k + 16 * (lane >> 4);
    v16bf b;
#pragma unroll
    for (int i = 0; i < 16; ++i) b[i] = p[i];
    return b;
}

// ---------------------------------------------------------------------------
// Generic GEMM: out[M,N] = act(A[M,Kp] @ W[N,Kp]^T + bias).
// 4 waves/WG; each wave owns TWO 16x16 N-tiles (nTiles must be EVEN, so the
// k-loop is branch-free). M fixed at 128 (grid.x = 8). Optional f32/bf16 out.
// ---------------------------------------------------------------------------
__global__ void wmma_gemm_kernel(const __bf16* __restrict__ A, int ldA,
                                 const __bf16* __restrict__ W, int ldW,
                                 const float* __restrict__ bias, int biasN,
                                 float* __restrict__ outF, int ldO,
                                 __bf16* __restrict__ outB, int ldOb,
                                 int kIters, int nTiles, int nStore, int relu) {
    int lane  = threadIdx.x & 31;
    int wave  = threadIdx.x >> 5;
    int mbase = blockIdx.x * 16;
    int t0    = (blockIdx.y * 4 + wave) * 2;
    if (t0 >= nTiles) return;                 // wave-uniform exit, before any WMMA
    int nb0 = t0 * 16, nb1 = nb0 + 16;

    v8f c0 = {}, c1 = {};
    for (int kk = 0; kk < kIters; ++kk) {
        int k = kk * 32;
        __builtin_prefetch(W + (size_t)(nb0 + (lane & 15)) * ldW + k + 64, 0, 3);
        v16bf a  = load_a_bf(A, ldA, mbase, k, lane);
        v16bf b0 = load_b_w(W, ldW, nb0, k, lane);
        c0 = __builtin_amdgcn_wmma_f32_16x16x32_bf16(false, a, false, b0,
                                                     (short)0, c0, false, false);
        v16bf b1 = load_b_w(W, ldW, nb1, k, lane);
        c1 = __builtin_amdgcn_wmma_f32_16x16x32_bf16(false, a, false, b1,
                                                     (short)0, c1, false, false);
    }
    int hi = lane >> 4, col = lane & 15;
#pragma unroll
    for (int tt = 0; tt < 2; ++tt) {
        int n = (tt ? nb1 : nb0) + col;
        v8f c = tt ? c1 : c0;
#pragma unroll
        for (int j = 0; j < 8; ++j) {
            int m = mbase + 8 * hi + j;
            if (n < nStore) {
                float v = c[j] + ((n < biasN) ? bias[n] : 0.f);
                if (relu) v = fmaxf(v, 0.f);
                if (outF) outF[(size_t)m * ldO + n]  = v;
                if (outB) outB[(size_t)m * ldOb + n] = (__bf16)v;
            }
        }
    }
}

// ---------------------------------------------------------------------------
// Fused GRU step: h' = (1-z)*tanh(in + r*hn) + z*h, all 6 gate GEMMs in WMMA.
// A-side operands (x, h) are bf16; f32 h is read only for the z*h term.
// Epilogue writes both the f32 and bf16 next-hidden.
// grid = (8, 8), block = 128 (4 waves), one 16x16 (batch x hidden) tile/wave.
// ---------------------------------------------------------------------------
__global__ void gru_step_kernel(const __bf16* __restrict__ xin, int ldx,
                                const __bf16* __restrict__ Wih, int kihIters,
                                const __bf16* __restrict__ Whh,
                                const float* __restrict__ bih,
                                const float* __restrict__ bhh,
                                const float* __restrict__ hinF,
                                const __bf16* __restrict__ hinB,
                                float* __restrict__ houtF,
                                __bf16* __restrict__ houtB) {
    int lane  = threadIdx.x & 31;
    int wave  = threadIdx.x >> 5;
    int mbase = blockIdx.x * 16;
    int nbase = (blockIdx.y * 4 + wave) * 16;
    int ldWih = kihIters * 32;

    v8f cir = {}, ciz = {}, cin = {}, chr_ = {}, chz = {}, chn = {};

    // input-side gates: gi = x @ Wih^T  (each A fragment feeds 3 WMMAs)
    for (int kk = 0; kk < kihIters; ++kk) {
        int k = kk * 32;
        v16bf a  = load_a_bf(xin, ldx, mbase, k, lane);
        v16bf br = load_b_w(Wih,                          ldWih, nbase, k, lane);
        v16bf bz = load_b_w(Wih + (size_t)H_ * ldWih,     ldWih, nbase, k, lane);
        v16bf bn = load_b_w(Wih + (size_t)2 * H_ * ldWih, ldWih, nbase, k, lane);
        cir = __builtin_amdgcn_wmma_f32_16x16x32_bf16(false, a, false, br, (short)0, cir, false, false);
        ciz = __builtin_amdgcn_wmma_f32_16x16x32_bf16(false, a, false, bz, (short)0, ciz, false, false);
        cin = __builtin_amdgcn_wmma_f32_16x16x32_bf16(false, a, false, bn, (short)0, cin, false, false);
    }
    // hidden-side gates: gh = h @ Whh^T (K = 512)
    for (int kk = 0; kk < 16; ++kk) {
        int k = kk * 32;
        __builtin_prefetch(Whh + (size_t)(nbase + (lane & 15)) * H_ + k + 64, 0, 3);
        v16bf a  = load_a_bf(hinB, H_, mbase, k, lane);
        v16bf br = load_b_w(Whh,                       H_, nbase, k, lane);
        v16bf bz = load_b_w(Whh + (size_t)H_ * H_,     H_, nbase, k, lane);
        v16bf bn = load_b_w(Whh + (size_t)2 * H_ * H_, H_, nbase, k, lane);
        chr_ = __builtin_amdgcn_wmma_f32_16x16x32_bf16(false, a, false, br, (short)0, chr_, false, false);
        chz  = __builtin_amdgcn_wmma_f32_16x16x32_bf16(false, a, false, bz, (short)0, chz,  false, false);
        chn  = __builtin_amdgcn_wmma_f32_16x16x32_bf16(false, a, false, bn, (short)0, chn,  false, false);
    }

    int hi = lane >> 4, col = lane & 15;
    int n  = nbase + col;
#pragma unroll
    for (int j = 0; j < 8; ++j) {
        int m = mbase + 8 * hi + j;
        float ir  = cir[j]  + bih[n];
        float iz  = ciz[j]  + bih[H_ + n];
        float in_ = cin[j]  + bih[2 * H_ + n];
        float hr  = chr_[j] + bhh[n];
        float hz  = chz[j]  + bhh[H_ + n];
        float hn  = chn[j]  + bhh[2 * H_ + n];
        float r  = sigmoidf_(ir + hr);
        float z  = sigmoidf_(iz + hz);
        float nn = tanh_fast_(in_ + r * hn);
        float ho = hinF[(size_t)m * H_ + n];
        float hv = (1.f - z) * nn + z * ho;
        houtF[(size_t)m * H_ + n] = hv;
        houtB[(size_t)m * H_ + n] = (__bf16)hv;
    }
}

// ---------------------------------------------------------------------------
// Small helper kernels
// ---------------------------------------------------------------------------
__global__ void cvt_pad_bf16_kernel(const float* __restrict__ src, int rows, int cols,
                                    __bf16* __restrict__ dst, int prows, int pcols) {
    int idx = blockIdx.x * blockDim.x + threadIdx.x;
    if (idx >= prows * pcols) return;
    int r = idx / pcols, c = idx % pcols;
    float v = (r < rows && c < cols) ? src[(size_t)r * cols + c] : 0.f;
    dst[idx] = (__bf16)v;
}

__global__ void zero_f32_kernel(float* __restrict__ p, int n) {
    int i = blockIdx.x * blockDim.x + threadIdx.x;
    if (i < n) p[i] = 0.f;
}
__global__ void zero_bf16_kernel(__bf16* __restrict__ p, int n) {
    int i = blockIdx.x * blockDim.x + threadIdx.x;
    if (i < n) p[i] = (__bf16)0.f;
}

// cat = [x_t(300) | hv(512) | zero-pad] per batch row, bf16, padded to 832
__global__ void build_cat_kernel(const float* __restrict__ xt, int xStride,
                                 const __bf16* __restrict__ hv,
                                 __bf16* __restrict__ cat) {
    int idx = blockIdx.x * blockDim.x + threadIdx.x;
    if (idx >= B_ * CATP_) return;
    int b = idx / CATP_, j = idx % CATP_;
    __bf16 v;
    if (j < D3_)       v = (__bf16)xt[(size_t)b * xStride + j];
    else if (j < CAT_) v = hv[(size_t)b * H_ + (j - D3_)];
    else               v = (__bf16)0.f;
    cat[idx] = v;
}

// softmax over TIN=256 per batch row; one block per row, LDS reductions
__global__ void softmax_tin_kernel(const float* __restrict__ logits,
                                   float* __restrict__ aw, int rowStrideOut) {
    __shared__ float red[TIN_];
    int b = blockIdx.x, t = threadIdx.x;
    float v = logits[(size_t)b * TIN_ + t];
    red[t] = v; __syncthreads();
    for (int s = TIN_ / 2; s > 0; s >>= 1) {
        if (t < s) red[t] = fmaxf(red[t], red[t + s]);
        __syncthreads();
    }
    float mx = red[0]; __syncthreads();
    float e = __expf(v - mx);
    red[t] = e; __syncthreads();
    for (int s = TIN_ / 2; s > 0; s >>= 1) {
        if (t < s) red[t] += red[t + s];
        __syncthreads();
    }
    aw[(size_t)b * rowStrideOut + t] = e / red[0];
}

// wctx[b,h] = sum_t aw[b,t] * encout_bf[t,b,h]  (bf16 reads, f32 accumulate)
__global__ void wctx_kernel(const float* __restrict__ aw, int awStride,
                            const __bf16* __restrict__ encoutB,
                            __bf16* __restrict__ wctxB) {
    int b = blockIdx.x;
    int h = blockIdx.y * 64 + threadIdx.x;
    const float* awb = aw + (size_t)b * awStride;
    float acc = 0.f;
    for (int t = 0; t < TIN_; ++t)
        acc += awb[t] * (float)encoutB[((size_t)t * B_ + b) * H_ + h];
    wctxB[(size_t)b * H_ + h] = (__bf16)acc;
}

// ---------------------------------------------------------------------------
// Host orchestration
// ---------------------------------------------------------------------------
static inline void cvt(const float* src, int r, int c, __bf16* dst, int pr, int pc,
                       hipStream_t s) {
    int n = pr * pc;
    cvt_pad_bf16_kernel<<<(n + 255) / 256, 256, 0, s>>>(src, r, c, dst, pr, pc);
}

extern "C" void kernel_launch(void* const* d_in, const int* in_sizes, int n_in,
                              void* d_out, int out_size, void* d_ws, size_t ws_size,
                              hipStream_t stream) {
    const float* x       = (const float*)d_in[0];
    const float* target  = (const float*)d_in[1];
    const float* encWih  = (const float*)d_in[2];
    const float* encWhh  = (const float*)d_in[3];
    const float* encbih  = (const float*)d_in[4];
    const float* encbhh  = (const float*)d_in[5];
    const float* decWih0 = (const float*)d_in[6];
    const float* decWih1 = (const float*)d_in[7];
    const float* decWhh  = (const float*)d_in[8];
    const float* decbih  = (const float*)d_in[9];
    const float* decbhh  = (const float*)d_in[10];
    const float* attnWw  = (const float*)d_in[11];
    const float* attnWb  = (const float*)d_in[12];
    const float* attnCw  = (const float*)d_in[13];
    const float* attnCb  = (const float*)d_in[14];
    const float* out1w   = (const float*)d_in[15];
    const float* out1b   = (const float*)d_in[16];
    const float* out2w   = (const float*)d_in[17];
    const float* out2b   = (const float*)d_in[18];
    float* out = (float*)d_out;
    (void)in_sizes; (void)n_in; (void)out_size; (void)ws_size;

    // ---- workspace carving ----
    char* ws = (char*)d_ws;
    size_t off = 0;
    auto take = [&](size_t bytes) -> char* {
        char* p = ws + off;
        off = (off + bytes + 255) & ~(size_t)255;
        return p;
    };
    const size_t BH = (size_t)B_ * H_;            // 65536 elements
    __bf16* encWih_bf  = (__bf16*)take((size_t)2 * THREEH_ * H_ * 2);
    __bf16* encWhh_bf  = (__bf16*)take((size_t)2 * THREEH_ * H_ * 2);
    __bf16* decWih0_bf = (__bf16*)take((size_t)THREEH_ * D3K_ * 2);
    __bf16* decWih1_bf = (__bf16*)take((size_t)THREEH_ * H_ * 2);
    __bf16* decWhh_bf  = (__bf16*)take((size_t)2 * THREEH_ * H_ * 2);
    __bf16* attnW_bf   = (__bf16*)take((size_t)TIN_ * CATP_ * 2);
    __bf16* attnC_bf   = (__bf16*)take((size_t)D3N_ * CATP_ * 2);   // N padded to 320
    __bf16* out1_bf    = (__bf16*)take((size_t)D3N_ * H_ * 2);
    __bf16* out2_bf    = (__bf16*)take((size_t)D3N_ * D3K_ * 2);
    __bf16* x_bf      = (__bf16*)take((size_t)B_ * TIN_ * IND_ * 2);
    float*  seq0      = (float*) take((size_t)TIN_ * BH * 4);  // layer-0 h chain (T,B,H)
    __bf16* seq0_bf   = (__bf16*)take((size_t)TIN_ * BH * 2);
    float*  encout    = (float*) take((size_t)TIN_ * BH * 4);  // layer-1 h chain (T,B,H)
    __bf16* encout_bf = (__bf16*)take((size_t)TIN_ * BH * 2);
    float*  zerosF    = (float*) take(BH * 4);
    __bf16* zerosB    = (__bf16*)take(BH * 2);
    float*  h0f[2] = { (float*)take(BH * 4), (float*)take(BH * 4) };
    float*  h1f[2] = { (float*)take(BH * 4), (float*)take(BH * 4) };
    __bf16* h0b[2] = { (__bf16*)take(BH * 2), (__bf16*)take(BH * 2) };
    __bf16* h1b[2] = { (__bf16*)take(BH * 2), (__bf16*)take(BH * 2) };
    __bf16* catbuf  = (__bf16*)take((size_t)B_ * CATP_ * 2);
    float*  logits  = (float*) take((size_t)B_ * TIN_ * 4);
    __bf16* wctxB   = (__bf16*)take(BH * 2);
    __bf16* xcbuf   = (__bf16*)take((size_t)B_ * D3K_ * 2);
    __bf16* o1buf   = (__bf16*)take((size_t)B_ * D3K_ * 2);

    // d_out layout: dec_out (B,SOUT,300) | dec_hidden (2,B,H) | attn (B,SOUT,TIN)
    float* decOut  = out;
    float* hidOut  = out + (size_t)B_ * SOUT_ * D3_;
    float* attnOut = hidOut + 2 * BH;

    // ---- zero buffers whose pad columns must stay 0 ----
    zero_f32_kernel<<<(int)((BH + 255) / 256), 256, 0, stream>>>(zerosF, (int)BH);
    zero_bf16_kernel<<<(int)((BH + 255) / 256), 256, 0, stream>>>(zerosB, (int)BH);
    zero_bf16_kernel<<<(B_ * D3K_ + 255) / 256, 256, 0, stream>>>(xcbuf, B_ * D3K_);
    zero_bf16_kernel<<<(B_ * D3K_ + 255) / 256, 256, 0, stream>>>(o1buf, B_ * D3K_);

    // ---- one-time f32 -> bf16 conversions (weights padded, x as-is) ----
    cvt(encWih,  2 * THREEH_, H_,   encWih_bf,  2 * THREEH_, H_,    stream);
    cvt(encWhh,  2 * THREEH_, H_,   encWhh_bf,  2 * THREEH_, H_,    stream);
    cvt(decWih0, THREEH_,     D3_,  decWih0_bf, THREEH_,     D3K_,  stream);
    cvt(decWih1, THREEH_,     H_,   decWih1_bf, THREEH_,     H_,    stream);
    cvt(decWhh,  2 * THREEH_, H_,   decWhh_bf,  2 * THREEH_, H_,    stream);
    cvt(attnWw,  TIN_,        CAT_, attnW_bf,   TIN_,        CATP_, stream);
    cvt(attnCw,  D3_,         CAT_, attnC_bf,   D3N_,        CATP_, stream);
    cvt(out1w,   D3_,         H_,   out1_bf,    D3N_,        H_,    stream);
    cvt(out2w,   D3_,         D3_,  out2_bf,    D3N_,        D3K_,  stream);
    cvt(x,       B_ * TIN_,   IND_, x_bf,       B_ * TIN_,   IND_,  stream);

    dim3 gruGrid(8, 8), gruBlk(128);
    // ---- encoder layer 0: x_bf(B,T,512) -> seq0(T,B,H) ----
    for (int t = 0; t < TIN_; ++t) {
        const __bf16* xin  = x_bf + (size_t)t * IND_;            // row stride T*IND
        const float*  hinF = t ? seq0    + (size_t)(t - 1) * BH : zerosF;
        const __bf16* hinB = t ? seq0_bf + (size_t)(t - 1) * BH : zerosB;
        gru_step_kernel<<<gruGrid, gruBlk, 0, stream>>>(
            xin, TIN_ * IND_, encWih_bf, IND_ / 32, encWhh_bf,
            encbih, encbhh, hinF, hinB,
            seq0 + (size_t)t * BH, seq0_bf + (size_t)t * BH);
    }
    // ---- encoder layer 1: seq0_bf -> encout ----
    const __bf16* encWih1 = encWih_bf + (size_t)THREEH_ * H_;
    const __bf16* encWhh1 = encWhh_bf + (size_t)THREEH_ * H_;
    for (int t = 0; t < TIN_; ++t) {
        const float*  hinF = t ? encout    + (size_t)(t - 1) * BH : zerosF;
        const __bf16* hinB = t ? encout_bf + (size_t)(t - 1) * BH : zerosB;
        gru_step_kernel<<<gruGrid, gruBlk, 0, stream>>>(
            seq0_bf + (size_t)t * BH, H_, encWih1, H_ / 32, encWhh1,
            encbih + THREEH_, encbhh + THREEH_, hinF, hinB,
            encout + (size_t)t * BH, encout_bf + (size_t)t * BH);
    }

    // ---- decoder init: hidden = encoder final hiddens (f32 + bf16 chains) ----
    hipMemcpyAsync(h0f[0], seq0      + (size_t)(TIN_ - 1) * BH, BH * 4, hipMemcpyDeviceToDevice, stream);
    hipMemcpyAsync(h1f[0], encout    + (size_t)(TIN_ - 1) * BH, BH * 4, hipMemcpyDeviceToDevice, stream);
    hipMemcpyAsync(h0b[0], seq0_bf   + (size_t)(TIN_ - 1) * BH, BH * 2, hipMemcpyDeviceToDevice, stream);
    hipMemcpyAsync(h1b[0], encout_bf + (size_t)(TIN_ - 1) * BH, BH * 2, hipMemcpyDeviceToDevice, stream);

    const __bf16* decWhh1 = decWhh_bf + (size_t)THREEH_ * H_;
    const int catElems = B_ * CATP_;
    int cur = 0;

    for (int s = 0; s < SOUT_; ++s) {
        int nxt = cur ^ 1;
        const float* xt = target + (size_t)s * D3_;              // row stride 33*300
        // attention weights: softmax(cat @ attnW^T + b)   (16 N-tiles, even)
        build_cat_kernel<<<(catElems + 255) / 256, 256, 0, stream>>>(
            xt, (SOUT_ + 1) * D3_, h1b[cur], catbuf);
        wmma_gemm_kernel<<<dim3(8, 2), 128, 0, stream>>>(
            catbuf, CATP_, attnW_bf, CATP_, attnWb, TIN_,
            logits, TIN_, (__bf16*)nullptr, 0,
            CATP_ / 32, TIN_ / 16, TIN_, 0);
        softmax_tin_kernel<<<B_, TIN_, 0, stream>>>(
            logits, attnOut + (size_t)s * TIN_, SOUT_ * TIN_);
        // weighted context over encoder outputs (bf16 reads)
        wctx_kernel<<<dim3(B_, H_ / 64), 64, 0, stream>>>(
            attnOut + (size_t)s * TIN_, SOUT_ * TIN_, encout_bf, wctxB);
        // xc = relu([x_t | wctx] @ attnC^T + b) -> bf16   (20 N-tiles, even)
        build_cat_kernel<<<(catElems + 255) / 256, 256, 0, stream>>>(
            xt, (SOUT_ + 1) * D3_, wctxB, catbuf);
        wmma_gemm_kernel<<<dim3(8, 3), 128, 0, stream>>>(
            catbuf, CATP_, attnC_bf, CATP_, attnCb, D3_,
            (float*)nullptr, 0, xcbuf, D3K_,
            CATP_ / 32, D3N_ / 16, D3S_, 1);
        // two GRU cells
        gru_step_kernel<<<gruGrid, gruBlk, 0, stream>>>(
            xcbuf, D3K_, decWih0_bf, D3K_ / 32, decWhh_bf,
            decbih, decbhh, h0f[cur], h0b[cur], h0f[nxt], h0b[nxt]);
        gru_step_kernel<<<gruGrid, gruBlk, 0, stream>>>(
            h0b[nxt], H_, decWih1_bf, H_ / 32, decWhh1,
            decbih + THREEH_, decbhh + THREEH_, h1f[cur], h1b[cur], h1f[nxt], h1b[nxt]);
        // output MLP: (h1 @ out1^T + b1) @ out2^T + b2 -> dec_out[:, s, :]
        wmma_gemm_kernel<<<dim3(8, 3), 128, 0, stream>>>(
            h1b[nxt], H_, out1_bf, H_, out1b, D3_,
            (float*)nullptr, 0, o1buf, D3K_,
            H_ / 32, D3N_ / 16, D3S_, 0);
        wmma_gemm_kernel<<<dim3(8, 3), 128, 0, stream>>>(
            o1buf, D3K_, out2_bf, D3K_, out2b, D3_,
            decOut + (size_t)s * D3_, SOUT_ * D3_, (__bf16*)nullptr, 0,
            D3K_ / 32, D3N_ / 16, D3_, 0);
        cur = nxt;
    }

    // ---- final decoder hidden -> d_out ----
    hipMemcpyAsync(hidOut,      h0f[cur], BH * 4, hipMemcpyDeviceToDevice, stream);
    hipMemcpyAsync(hidOut + BH, h1f[cur], BH * 4, hipMemcpyDeviceToDevice, stream);
}
